// ECGCGNN_1211180778318
// MI455X (gfx1250) — compile-verified, matrix operand order checked by reading, not run on hardware
//
#include <hip/hip_runtime.h>
#include <hip/hip_bf16.h>

#define NUM_NODES 12288
#define NUM_GRAPHS 1024
#define LEADS 12
#define SIG_LEN 2048
#define NUM_EDGES 196608

typedef __attribute__((ext_vector_type(16))) _Float16 v16h;
typedef __attribute__((ext_vector_type(8)))  _Float16 v8h;
typedef __attribute__((ext_vector_type(8)))  float    v8f;

#define CAT16(lo, hi) __builtin_shufflevector((lo), (hi), 0,1,2,3,4,5,6,7,8,9,10,11,12,13,14,15)

// ---------------------------------------------------------------------------
// Kernel 1: fused per-node frontend.
//   conv1 (1->16ch, k7 s2 p3, relu, fp32 VALU)  -> f16 in LDS, layout
//       h1t[pos+2][ic]  (position-major, 2 zero guard rows each side)
//   conv2 (16->32ch, k5 s2 p2) as WMMA GEMM with K re-ordered k = tap*16+ic
//       -> every B fragment half is 8 contiguous f16 => ds_load_b128
//   relu + mean over 512 positions (register acc, one shfl reduce per wave)
//   linear 32->64  -> h0 (f16)
// ---------------------------------------------------------------------------
__global__ __launch_bounds__(256) void ecg_frontend_kernel(
    const float* __restrict__ x_raw,
    const float* __restrict__ w1, const float* __restrict__ b1,
    const float* __restrict__ w2, const float* __restrict__ b2,
    const float* __restrict__ lw, const float* __restrict__ lb,
    _Float16* __restrict__ h0f16)
{
    __shared__ float    xs[SIG_LEN];          // 8 KB
    __shared__ _Float16 h1t[1028 * 16];       // 32.1 KB, [pos+2][ic]
    __shared__ float    w1s[112];
    __shared__ float    b1s[16];
    __shared__ float    csum[32];
    __shared__ float    hm[32];

    const int node = blockIdx.x;
    const int tid  = threadIdx.x;
    const int lane = tid & 31;
    const int wave = tid >> 5;
    const int hi   = lane >> 4;
    const int l16  = lane & 15;

    // stage signal + conv1 weights; zero guard rows (pos rows 0,1,1026,1027)
    const float4* xin = (const float4*)(x_raw + (size_t)node * SIG_LEN);
    float4* xs4 = (float4*)xs;
    for (int i = tid; i < SIG_LEN / 4; i += 256) xs4[i] = xin[i];
    if (tid < 112) w1s[tid] = w1[tid];
    if (tid < 16)  b1s[tid] = b1[tid];
    if (tid < 32)  csum[tid] = 0.f;
    if (tid < 64) {
        const int r = tid >> 4;
        const int row = (r < 2) ? r : (r + 1024);
        h1t[row * 16 + (tid & 15)] = (_Float16)0.f;
    }
    __syncthreads();

    // conv1: 1024 positions x 16 ch; consecutive threads -> consecutive ic
    for (int i = tid; i < 16 * 1024; i += 256) {
        const int p = i >> 4, oc = i & 15;
        float acc = b1s[oc];
        const int ip0 = 2 * p - 3;
#pragma unroll
        for (int t = 0; t < 7; ++t) {
            const int ip = ip0 + t;
            if (ip >= 0 && ip < SIG_LEN) acc += xs[ip] * w1s[oc * 7 + t];
        }
        h1t[(p + 2) * 16 + oc] = (_Float16)(acc > 0.f ? acc : 0.f);
    }
    __syncthreads();

    // ---- conv2 as WMMA: A = weights 32 x 96 (k = tap*16 + ic), B = im2col
    // A fragments (built once, scalar global loads of the tiny weight tensor)
    v16h afrag[2][3];
#pragma unroll
    for (int mt = 0; mt < 2; ++mt)
#pragma unroll
        for (int ks = 0; ks < 3; ++ks) {
            const int m = mt * 16 + l16;
            v16h a;
#pragma unroll
            for (int e = 0; e < 16; ++e) {
                const int k = ((e < 8) ? e : e + 8) + hi * 8 + ks * 32;
                _Float16 val = (_Float16)0.f;
                if (k < 80) {
                    const int tap = k >> 4, ic = k & 15;
                    val = (_Float16)w2[m * 80 + ic * 5 + tap];
                }
                a[e] = val;
            }
            afrag[mt][ks] = a;
        }

    // bias per accumulator slot (C layout: lane l16 = position, v+8*hi = ch)
    float bias[2][8];
#pragma unroll
    for (int mt = 0; mt < 2; ++mt)
#pragma unroll
        for (int v = 0; v < 8; ++v) bias[mt][v] = b2[mt * 16 + v + 8 * hi];

    float sacc[2][8];
#pragma unroll
    for (int mt = 0; mt < 2; ++mt)
#pragma unroll
        for (int v = 0; v < 8; ++v) sacc[mt][v] = 0.f;

    // each wave covers 4 N-tiles of 16 positions (8 waves * 4 * 16 = 512)
    for (int q = 0; q < 4; ++q) {
        const int n = (wave * 4 + q) * 16 + l16;   // output position
        v16h bfr[3];
#pragma unroll
        for (int ks = 0; ks < 3; ++ks) {
            // run 1: k = 32*ks + 8*hi + 0..7  (tap = k>>4, ic run of 8)
            const int kA = 32 * ks + 8 * hi;
            const int rowA = 2 * n + (kA >> 4);          // = ip + 2, in bounds
            v8h lo = *(const v8h*)(h1t + rowA * 16 + (kA & 15));
            v8h hh = {};
            if (ks < 2) {                                 // k >= 80 -> zeros
                const int kB = kA + 16;
                const int rowB = 2 * n + (kB >> 4);
                hh = *(const v8h*)(h1t + rowB * 16 + (kB & 15));
            }
            bfr[ks] = CAT16(lo, hh);
        }
#pragma unroll
        for (int mt = 0; mt < 2; ++mt) {
            v8f c;
#pragma unroll
            for (int v = 0; v < 8; ++v) c[v] = bias[mt][v];
            c = __builtin_amdgcn_wmma_f32_16x16x32_f16(false, afrag[mt][0],
                    false, bfr[0], (short)0, c, false, false);
            c = __builtin_amdgcn_wmma_f32_16x16x32_f16(false, afrag[mt][1],
                    false, bfr[1], (short)0, c, false, false);
            c = __builtin_amdgcn_wmma_f32_16x16x32_f16(false, afrag[mt][2],
                    false, bfr[2], (short)0, c, false, false);
#pragma unroll
            for (int v = 0; v < 8; ++v)
                sacc[mt][v] += (c[v] > 0.f ? c[v] : 0.f);   // relu + pos sum
        }
    }

    // one reduction per wave: sum the 16 positions held across lanes
#pragma unroll
    for (int mt = 0; mt < 2; ++mt)
#pragma unroll
        for (int v = 0; v < 8; ++v) {
            float r = sacc[mt][v];
            r += __shfl_xor(r, 1, 32);
            r += __shfl_xor(r, 2, 32);
            r += __shfl_xor(r, 4, 32);
            r += __shfl_xor(r, 8, 32);
            if (l16 == 0) atomicAdd(&csum[mt * 16 + v + 8 * hi], r);
        }
    __syncthreads();

    if (tid < 32) hm[tid] = csum[tid] * (1.f / 512.f);
    __syncthreads();

    // linear 32 -> 64
    if (tid < 64) {
        float acc = lb[tid];
#pragma unroll
        for (int c = 0; c < 32; ++c) acc += hm[c] * lw[c * 64 + tid];
        h0f16[(size_t)node * 64 + tid] = (_Float16)acc;
    }
}

// ---------------------------------------------------------------------------
// Weight transpose + f32->f16: W[K x M] -> Wt[M x K]
// ---------------------------------------------------------------------------
__global__ void ecg_wcast_kernel(const float* __restrict__ W,
                                 _Float16* __restrict__ Wt, int K, int M)
{
    const int i = blockIdx.x * 256 + threadIdx.x;
    if (i < K * M) {
        const int k = i / M, m = i % M;
        Wt[m * K + k] = (_Float16)W[i];
    }
}

// ---------------------------------------------------------------------------
// GEMM: Y[N x M] = X[N x K] (f16) * W[K x M] (Wt = f16, M x K), f32 out.
// One wave per 16x16 tile. Fragment halves are contiguous 8xf16 runs at
// kb+8*hi and kb+16+8*hi -> two global_load_b128 per matrix per K-step.
// ---------------------------------------------------------------------------
__global__ __launch_bounds__(256) void ecg_gemm_kernel(
    const _Float16* __restrict__ X, const _Float16* __restrict__ Wt,
    float* __restrict__ Y, int K, int M)
{
    const int lane  = threadIdx.x & 31;
    const int mtile = threadIdx.x >> 5;
    if (mtile >= (M >> 4)) return;
    const int ntile = blockIdx.x;
    const int hi = lane >> 4, l16 = lane & 15;
    const _Float16* ap = X  + (size_t)(ntile * 16 + l16) * K + 8 * hi;
    const _Float16* bp = Wt + (size_t)(mtile * 16 + l16) * K + 8 * hi;

    v8f c = {};
    for (int kb = 0; kb < K; kb += 32) {
        const v8h alo = *(const v8h*)(ap + kb);
        const v8h ahi = *(const v8h*)(ap + kb + 16);
        const v8h blo = *(const v8h*)(bp + kb);
        const v8h bhi = *(const v8h*)(bp + kb + 16);
        c = __builtin_amdgcn_wmma_f32_16x16x32_f16(false, CAT16(alo, ahi),
                false, CAT16(blo, bhi), (short)0, c, false, false);
    }
#pragma unroll
    for (int v = 0; v < 8; ++v)
        Y[(size_t)(ntile * 16 + v + 8 * hi) * M + mtile * 16 + l16] = c[v];
}

// ---------------------------------------------------------------------------
// GCN support kernels
// ---------------------------------------------------------------------------
__global__ void ecg_deg_init_kernel(float* __restrict__ deg, int n)
{
    const int i = blockIdx.x * 256 + threadIdx.x;
    if (i < n) deg[i] = 1.f;               // self loop
}

__global__ void ecg_deg_edge_kernel(const int* __restrict__ dst,
                                    float* __restrict__ deg, int ne)
{
    const int i = blockIdx.x * 256 + threadIdx.x;
    if (i < ne) atomicAdd(&deg[dst[i]], 1.f);
}

__global__ void ecg_dis_kernel(const float* __restrict__ deg,
                               float* __restrict__ dis, int n)
{
    const int i = blockIdx.x * 256 + threadIdx.x;
    if (i < n) dis[i] = rsqrtf(deg[i]);
}

// acc[i][c] = hw[i][c] * dis[i]^2   (self-loop contribution, M = 128)
__global__ void ecg_self_init_kernel(const float* __restrict__ hw,
                                     const float* __restrict__ dis,
                                     float* __restrict__ acc, int n)
{
    const int i = blockIdx.x * 256 + threadIdx.x;
    if (i < n) {
        const float d = dis[i >> 7];
        acc[i] = hw[i] * d * d;
    }
}

// one wave per edge, lanes stride the 128 channels (coalesced)
__global__ void ecg_scatter_kernel(const int* __restrict__ src,
                                   const int* __restrict__ dst,
                                   const float* __restrict__ hw,
                                   const float* __restrict__ dis,
                                   float* __restrict__ acc, int ne)
{
    const int gid  = blockIdx.x * blockDim.x + threadIdx.x;
    const int e    = gid >> 5;
    const int lane = gid & 31;
    if (e >= ne) return;
    const int s = src[e], d = dst[e];
    const float norm = dis[s] * dis[d];
    const float* hs = hw  + (size_t)s * 128;
    float*       ad = acc + (size_t)d * 128;
#pragma unroll
    for (int c = lane; c < 128; c += 32)
        atomicAdd(&ad[c], hs[c] * norm);
}

// out = relu(acc + bias); optionally emit f32 and/or f16 copies
__global__ void ecg_finalize_kernel(const float* __restrict__ acc,
                                    const float* __restrict__ bias,
                                    float* __restrict__ out32,
                                    _Float16* __restrict__ out16, int n)
{
    const int i = blockIdx.x * 256 + threadIdx.x;
    if (i < n) {
        float v = acc[i] + bias[i & 127];
        v = v > 0.f ? v : 0.f;
        if (out32) out32[i] = v;
        if (out16) out16[i] = (_Float16)v;
    }
}

// mean-pool 12 nodes per graph + FC 128 -> 5
__global__ __launch_bounds__(128) void ecg_pool_fc_kernel(
    const float* __restrict__ h2, const float* __restrict__ fcw,
    const float* __restrict__ fcb, float* __restrict__ out)
{
    __shared__ float p[128];
    const int g = blockIdx.x, c = threadIdx.x;
    float s = 0.f;
#pragma unroll
    for (int i = 0; i < LEADS; ++i)
        s += h2[(size_t)(g * LEADS + i) * 128 + c];
    p[c] = s * (1.f / (float)LEADS);
    __syncthreads();
    if (c < 5) {
        float acc = fcb[c];
#pragma unroll 16
        for (int k = 0; k < 128; ++k) acc += p[k] * fcw[k * 5 + c];
        out[g * 5 + c] = acc;
    }
}

// ---------------------------------------------------------------------------
extern "C" void kernel_launch(void* const* d_in, const int* in_sizes, int n_in,
                              void* d_out, int out_size, void* d_ws, size_t ws_size,
                              hipStream_t stream)
{
    const float* x_raw  = (const float*)d_in[0];
    const float* c1w    = (const float*)d_in[1];
    const float* c1b    = (const float*)d_in[2];
    const float* c2w    = (const float*)d_in[3];
    const float* c2b    = (const float*)d_in[4];
    const float* lin_w  = (const float*)d_in[5];
    const float* lin_b  = (const float*)d_in[6];
    const float* g1w    = (const float*)d_in[7];
    const float* g1b    = (const float*)d_in[8];
    const float* g2w    = (const float*)d_in[9];
    const float* g2b    = (const float*)d_in[10];
    const float* fcw    = (const float*)d_in[11];
    const float* fcb    = (const float*)d_in[12];
    const int*   eidx   = (const int*)d_in[13];
    const int*   src    = eidx;
    const int*   dst    = eidx + NUM_EDGES;
    float*       out    = (float*)d_out;

    // workspace carve-up (256B aligned)
    char* ws = (char*)d_ws;
    size_t off = 0;
    auto carve = [&](size_t bytes) { char* p = ws + off;
        off = (off + bytes + 255) & ~(size_t)255; return p; };
    _Float16* h0f16 = (_Float16*)carve((size_t)NUM_NODES * 64 * 2);
    _Float16* wt1   = (_Float16*)carve(128 * 64 * 2);
    _Float16* wt2   = (_Float16*)carve(128 * 128 * 2);
    float*    hw    = (float*)carve((size_t)NUM_NODES * 128 * 4);
    float*    acc   = (float*)carve((size_t)NUM_NODES * 128 * 4);
    _Float16* h1f16 = (_Float16*)carve((size_t)NUM_NODES * 128 * 2);
    float*    h2f32 = (float*)carve((size_t)NUM_NODES * 128 * 4);
    float*    deg   = (float*)carve((size_t)NUM_NODES * 4);
    float*    dis   = (float*)carve((size_t)NUM_NODES * 4);
    (void)ws_size; (void)in_sizes; (void)n_in; (void)out_size;

    const int NC = NUM_NODES * 128;                 // node-channel count

    // frontend: conv1 + WMMA conv2 + mean + linear
    ecg_frontend_kernel<<<NUM_NODES, 256, 0, stream>>>(
        x_raw, c1w, c1b, c2w, c2b, lin_w, lin_b, h0f16);

    // weight transpose/convert
    ecg_wcast_kernel<<<(64 * 128 + 255) / 256, 256, 0, stream>>>(g1w, wt1, 64, 128);
    ecg_wcast_kernel<<<(128 * 128 + 255) / 256, 256, 0, stream>>>(g2w, wt2, 128, 128);

    // degrees / normalization
    ecg_deg_init_kernel<<<(NUM_NODES + 255) / 256, 256, 0, stream>>>(deg, NUM_NODES);
    ecg_deg_edge_kernel<<<(NUM_EDGES + 255) / 256, 256, 0, stream>>>(dst, deg, NUM_EDGES);
    ecg_dis_kernel<<<(NUM_NODES + 255) / 256, 256, 0, stream>>>(deg, dis, NUM_NODES);

    // GCN layer 1: XW (WMMA), self term, edge scatter, bias+relu -> f16
    ecg_gemm_kernel<<<NUM_NODES / 16, 256, 0, stream>>>(h0f16, wt1, hw, 64, 128);
    ecg_self_init_kernel<<<NC / 256, 256, 0, stream>>>(hw, dis, acc, NC);
    ecg_scatter_kernel<<<(NUM_EDGES * 32) / 256, 256, 0, stream>>>(src, dst, hw, dis, acc, NUM_EDGES);
    ecg_finalize_kernel<<<NC / 256, 256, 0, stream>>>(acc, g1b, nullptr, h1f16, NC);

    // GCN layer 2
    ecg_gemm_kernel<<<NUM_NODES / 16, 256, 0, stream>>>(h1f16, wt2, hw, 128, 128);
    ecg_self_init_kernel<<<NC / 256, 256, 0, stream>>>(hw, dis, acc, NC);
    ecg_scatter_kernel<<<(NUM_EDGES * 32) / 256, 256, 0, stream>>>(src, dst, hw, dis, acc, NUM_EDGES);
    ecg_finalize_kernel<<<NC / 256, 256, 0, stream>>>(acc, g2b, h2f32, nullptr, NC);

    // pooling + classifier
    ecg_pool_fc_kernel<<<NUM_GRAPHS, 128, 0, stream>>>(h2f32, fcw, fcb, out);
}
